// AutoEncoder_57363583205482
// MI455X (gfx1250) — matrix operand order repaired and checked by reading, not compiled, not run on hardware
//
#include <hip/hip_runtime.h>
#include <hip/hip_bf16.h>

typedef __attribute__((ext_vector_type(16))) _Float16 v16h;
typedef __attribute__((ext_vector_type(8)))  float    v8f;
typedef __attribute__((ext_vector_type(4)))  unsigned v4u;
typedef __attribute__((ext_vector_type(8)))  int      v8i;
typedef __attribute__((ext_vector_type(4)))  int      v4i;

#define NNODES 819200
#define NEDGES 6553600
#define NGRAPH 16384
#define NPG    50
#define NPAIR  1225
#define NOUTC  2450   // 2*NPAIR
#define NCPAD  2464   // NOUTC padded to 16 (zero columns)

// ---------------------------------------------------------------------------
// Edge scatter-add, layer1 (K=10, 40B rows, unaligned -> scalar).
// ---------------------------------------------------------------------------
__global__ void k_scatter10(const int* __restrict__ ei,
                            const float* __restrict__ F,
                            float* __restrict__ agg)
{
    long e = (long)blockIdx.x * blockDim.x + threadIdx.x;
    int s = ei[e];
    int t = ei[(long)NEDGES + e];
    const float* fr = F + (long)s * 10;
    float* ar = agg + (long)t * 32;
#pragma unroll
    for (int k = 0; k < 10; ++k) atomicAdd(&ar[k], fr[k]);
}

// ---------------------------------------------------------------------------
// Edge scatter-add, layer2: 4 threads/edge, 2x b128 loads + 8 f32 atomics each.
// The 105MB node table fits in the 192MB L2, so atomics resolve in L2.
// ---------------------------------------------------------------------------
__global__ void k_scatter32(const int* __restrict__ ei,
                            const float* __restrict__ F,
                            float* __restrict__ agg)
{
    long t = (long)blockIdx.x * blockDim.x + threadIdx.x;
    long e = t >> 2;
    int  q = (int)(t & 3);
    int s  = ei[e];
    int tg = ei[(long)NEDGES + e];
    const float4* fr = (const float4*)(F + (long)s * 32) + q * 2;
    float4 a0 = fr[0];
    float4 a1 = fr[1];
    float* ar = agg + (long)tg * 32 + q * 8;
    atomicAdd(ar + 0, a0.x); atomicAdd(ar + 1, a0.y);
    atomicAdd(ar + 2, a0.z); atomicAdd(ar + 3, a0.w);
    atomicAdd(ar + 4, a1.x); atomicAdd(ar + 5, a1.y);
    atomicAdd(ar + 6, a1.z); atomicAdd(ar + 7, a1.w);
}

// ---------------------------------------------------------------------------
// Fused node MLP tile: Y[16,32] = relu( T(X)[16,32] @ W[KIN,32] + b )
// T is (X + AGG) or BN-affine (a*x+c). Optional per-feature sum/sumsq stats.
// One wave = one 16-row tile; 2x v_wmma_f32_16x16x32_f16 per wave.
// ---------------------------------------------------------------------------
template<int KIN, bool USE_AGG, bool USE_BN, bool STATS>
__global__ void k_fused_gemm(const float* __restrict__ X, int ldx,
                             const float* __restrict__ AGG,   // stride 32
                             const float* __restrict__ bnp,   // a[32], c[32]
                             const float* __restrict__ W,     // [KIN][32] row-major
                             const float* __restrict__ bias,  // [32]
                             float* __restrict__ Y,           // [rows][32]
                             float* __restrict__ stats)       // sum[32], sumsq[32]
{
    __shared__ float sred[64];
    const int tid  = threadIdx.x;
    const int lane = tid & 31;
    const int wave = tid >> 5;
    const int half = (lane < 16) ? 0 : 1;
    const int r    = lane & 15;
    const long rowbase = (long)blockIdx.x * 128 + (long)wave * 16;
    const long ridx    = rowbase + r;

    if (STATS) {
        if (tid < 64) sred[tid] = 0.0f;
        __syncthreads();
    }

    // A tile: lane<16 holds row r, K in {0..7}u{16..23}; lane>=16: {8..15}u{24..31}
    float va[16];
    if constexpr (KIN == 32) {
        const float4* xr = (const float4*)(X + ridx * 32);
        float4 c0 = xr[half * 2 + 0];
        float4 c1 = xr[half * 2 + 1];
        float4 c2 = xr[half * 2 + 4];
        float4 c3 = xr[half * 2 + 5];
        if constexpr (USE_AGG) {
            const float4* ar = (const float4*)(AGG + ridx * 32);
            float4 g0 = ar[half * 2 + 0], g1 = ar[half * 2 + 1];
            float4 g2 = ar[half * 2 + 4], g3 = ar[half * 2 + 5];
            c0.x += g0.x; c0.y += g0.y; c0.z += g0.z; c0.w += g0.w;
            c1.x += g1.x; c1.y += g1.y; c1.z += g1.z; c1.w += g1.w;
            c2.x += g2.x; c2.y += g2.y; c2.z += g2.z; c2.w += g2.w;
            c3.x += g3.x; c3.y += g3.y; c3.z += g3.z; c3.w += g3.w;
        }
        if constexpr (USE_BN) {
            const float4* pa = (const float4*)bnp;
            const float4* pc = (const float4*)(bnp + 32);
            float4 s0 = pa[half * 2 + 0], s1 = pa[half * 2 + 1];
            float4 s2 = pa[half * 2 + 4], s3 = pa[half * 2 + 5];
            float4 t0 = pc[half * 2 + 0], t1 = pc[half * 2 + 1];
            float4 t2 = pc[half * 2 + 4], t3 = pc[half * 2 + 5];
            c0.x = c0.x * s0.x + t0.x; c0.y = c0.y * s0.y + t0.y;
            c0.z = c0.z * s0.z + t0.z; c0.w = c0.w * s0.w + t0.w;
            c1.x = c1.x * s1.x + t1.x; c1.y = c1.y * s1.y + t1.y;
            c1.z = c1.z * s1.z + t1.z; c1.w = c1.w * s1.w + t1.w;
            c2.x = c2.x * s2.x + t2.x; c2.y = c2.y * s2.y + t2.y;
            c2.z = c2.z * s2.z + t2.z; c2.w = c2.w * s2.w + t2.w;
            c3.x = c3.x * s3.x + t3.x; c3.y = c3.y * s3.y + t3.y;
            c3.z = c3.z * s3.z + t3.z; c3.w = c3.w * s3.w + t3.w;
        }
        va[0] = c0.x; va[1]  = c0.y; va[2]  = c0.z; va[3]  = c0.w;
        va[4] = c1.x; va[5]  = c1.y; va[6]  = c1.z; va[7]  = c1.w;
        va[8] = c2.x; va[9]  = c2.y; va[10] = c2.z; va[11] = c2.w;
        va[12] = c3.x; va[13] = c3.y; va[14] = c3.z; va[15] = c3.w;
    } else {
#pragma unroll
        for (int i = 0; i < 16; ++i) {
            int k = (i & 7) + half * 8 + ((i & 8) ? 16 : 0);
            float v = 0.0f;
            if (k < KIN) {
                v = X[ridx * (long)ldx + k];
                if (USE_AGG) v += AGG[ridx * 32 + k];
                if (USE_BN)  v = v * bnp[k] + bnp[32 + k];
            }
            va[i] = v;
        }
    }
    v16h a;
#pragma unroll
    for (int i = 0; i < 16; ++i) a[i] = (_Float16)va[i];

    // B tiles: lane<16 holds column (r [+16]) with K in {0..7}u{16..23}; lane>=16: {8..15}u{24..31}
    v16h b0, b1;
#pragma unroll
    for (int i = 0; i < 16; ++i) {
        int k = (i & 7) + half * 8 + ((i & 8) ? 16 : 0);
        float w0 = 0.0f, w1 = 0.0f;
        if (k < KIN) { w0 = W[k * 32 + r]; w1 = W[k * 32 + r + 16]; }
        b0[i] = (_Float16)w0;
        b1[i] = (_Float16)w1;
    }

    // C = bias broadcast down rows (lane owns one output column)
    v8f c0, c1;
    float bb0 = bias[r], bb1 = bias[r + 16];
#pragma unroll
    for (int i = 0; i < 8; ++i) { c0[i] = bb0; c1[i] = bb1; }

    v8f d0 = __builtin_amdgcn_wmma_f32_16x16x32_f16(false, a, false, b0, (short)0, c0, false, false);
    v8f d1 = __builtin_amdgcn_wmma_f32_16x16x32_f16(false, a, false, b1, (short)0, c1, false, false);

    float s0 = 0.f, q0 = 0.f, s1 = 0.f, q1 = 0.f;
#pragma unroll
    for (int i = 0; i < 8; ++i) {
        int  m  = i + half * 8;
        long rr = rowbase + m;
        float y0 = d0[i] > 0.0f ? d0[i] : 0.0f;
        float y1 = d1[i] > 0.0f ? d1[i] : 0.0f;
        Y[rr * 32 + r]      = y0;
        Y[rr * 32 + r + 16] = y1;
        if (STATS) { s0 += y0; q0 += y0 * y0; s1 += y1; q1 += y1 * y1; }
    }

    if (STATS) {
        // lanes L and L+16 hold the same column, different rows
        s0 += __shfl_xor(s0, 16, 32);  q0 += __shfl_xor(q0, 16, 32);
        s1 += __shfl_xor(s1, 16, 32);  q1 += __shfl_xor(q1, 16, 32);
        if (half == 0) {
            atomicAdd(&sred[r],           s0);
            atomicAdd(&sred[r + 16],      s1);
            atomicAdd(&sred[32 + r],      q0);
            atomicAdd(&sred[32 + r + 16], q1);
        }
        __syncthreads();
        if (tid < 64) atomicAdd(&stats[tid], sred[tid]);
    }
}

// ---------------------------------------------------------------------------
// BN finalize: a = gamma*rsqrt(var+eps), c = beta - mu*a
// ---------------------------------------------------------------------------
__global__ void k_bnfin(const float* __restrict__ stats,
                        const float* __restrict__ gamma,
                        const float* __restrict__ beta,
                        float* __restrict__ bnp, float invN)
{
    int k = threadIdx.x;           // 32 threads
    float mu  = stats[k] * invN;
    float var = stats[32 + k] * invN - mu * mu;
    float a   = gamma[k] * rsqrtf(var + 1e-5f);
    bnp[k]      = a;
    bnp[32 + k] = beta[k] - mu * a;
}

// ---------------------------------------------------------------------------
// Graph pooling: xg[g,f] = sum of 50 consecutive node rows (coalesced 128B rows)
// ---------------------------------------------------------------------------
__global__ void k_graphsum(const float* __restrict__ H, float* __restrict__ XG)
{
    long t = (long)blockIdx.x * blockDim.x + threadIdx.x;  // NGRAPH*32 threads
    long g = t >> 5;
    int  f = (int)(t & 31);
    const float* p = H + g * (long)(NPG * 32) + f;
    float s = 0.0f;
#pragma unroll
    for (int r = 0; r < NPG; ++r) s += p[r * 32];
    XG[g * 32 + f] = s;
}

// ---------------------------------------------------------------------------
// Build f16 transposed+padded decoder weights: Wh[col][k], col in [0,2464)
// ---------------------------------------------------------------------------
__global__ void k_cvtW1(const float* __restrict__ W, _Float16* __restrict__ Wh)
{
    int t = blockIdx.x * blockDim.x + threadIdx.x;   // NCPAD*32 threads
    int col = t >> 5, k = t & 31;
    float v = (col < NOUTC) ? W[k * NOUTC + col] : 0.0f;
    Wh[col * 32 + k] = (_Float16)v;
}

// ---------------------------------------------------------------------------
// Fused decoder: logits = d @ dec_W1 + b + gumbel, pairwise hard argmax,
// scatter 0/1 into symmetric adjacency. B matrix (f16, transposed, 157KB)
// staged into LDS once per block via the Tensor Data Mover; pair LUT in LDS.
// One wave = 16 graphs; one WMMA per 16-col tile of the 2450 outputs.
// ---------------------------------------------------------------------------
__global__ void k_decode(const float* __restrict__ D,       // [B][32]
                         const _Float16* __restrict__ W1h,  // [2464][32] f16
                         const float* __restrict__ bias,    // [2450]
                         const float* __restrict__ gum,     // [B][1225][2]
                         float* __restrict__ adj)           // [B][50][50]
{
    __shared__ __align__(64) _Float16 shW[NCPAD * 32];   // 157,696 B
    __shared__ int sh_pair[NPAIR];

    const int tid  = threadIdx.x;
    const int lane = tid & 31;
    const int wave = tid >> 5;
    const int half = (lane < 16) ? 0 : 1;
    const int r    = lane & 15;
    const long g0  = (long)blockIdx.x * 128 + (long)wave * 16;

    // pair -> (iu, ju) lookup table, built once per block
    for (int p = tid; p < NPAIR; p += 256) {
        int i0 = (int)((99.0f - sqrtf((float)(9801 - 8 * p))) * 0.5f);
        if (i0 < 0) i0 = 0;
        if (i0 > 48) i0 = 48;
        while (i0 > 0 && (i0 * (99 - i0)) / 2 > p) --i0;
        while (((i0 + 1) * (98 - i0)) / 2 <= p) ++i0;
        sh_pair[p] = (i0 << 8) | (p - (i0 * (99 - i0)) / 2 + i0 + 1);
    }

#if __has_builtin(__builtin_amdgcn_tensor_load_to_lds) && __has_builtin(__builtin_amdgcn_s_wait_tensorcnt)
    // TDM: one wave issues a single 2D tensor load (32 x 2464 of 2B elements)
    if (tid < 32) {
        unsigned lds_addr = (unsigned)(size_t)(void*)&shW[0];
        unsigned long long ga = (unsigned long long)(size_t)W1h;
        v4u g0d;
        g0d[0] = 1u;                                                // count=1, user mode
        g0d[1] = lds_addr;                                          // lds_addr [63:32]
        g0d[2] = (unsigned)ga;                                      // global_addr lo
        g0d[3] = (unsigned)((ga >> 32) & 0x01FFFFFFull) | (2u << 30); // addr hi | type=2
        v8i g1d;
        g1d[0] = 0x00010000;          // workgroup_mask=0, data_size=1 (2 bytes)
        g1d[1] = (int)(32u << 16);    // tensor_dim0 = 32
        g1d[2] = (int)(2464u << 16);  // tensor_dim1 = 2464
        g1d[3] = (int)(32u << 16);    // tile_dim0 = 32
        g1d[4] = 2464;                // tile_dim1 = 2464
        g1d[5] = 32;                  // tensor_dim0_stride = 32
        g1d[6] = 0;
        g1d[7] = 0;
        v4i z4 = {0, 0, 0, 0};
        v8i z8 = {0, 0, 0, 0, 0, 0, 0, 0};
        __builtin_amdgcn_tensor_load_to_lds(g0d, g1d, z4, z4, z8, 0);
        __builtin_amdgcn_s_wait_tensorcnt(0);
    }
#else
    for (int c = tid; c < (NCPAD * 32) / 8; c += 256)
        ((float4*)shW)[c] = ((const float4*)W1h)[c];
#endif
    __syncthreads();

    // A tile: 16 graphs x 32 features, vectorized b128 loads
    const float4* dr = (const float4*)(D + (g0 + r) * 32);
    float4 a0 = dr[half * 4 + 0], a1 = dr[half * 4 + 1];
    float4 a2 = dr[half * 4 + 2], a3 = dr[half * 4 + 3];
    v16h a;
    a[0]  = (_Float16)a0.x; a[1]  = (_Float16)a0.y; a[2]  = (_Float16)a0.z; a[3]  = (_Float16)a0.w;
    a[4]  = (_Float16)a1.x; a[5]  = (_Float16)a1.y; a[6]  = (_Float16)a1.z; a[7]  = (_Float16)a1.w;
    a[8]  = (_Float16)a2.x; a[9]  = (_Float16)a2.y; a[10] = (_Float16)a2.z; a[11] = (_Float16)a2.w;
    a[12] = (_Float16)a3.x; a[13] = (_Float16)a3.y; a[14] = (_Float16)a3.z; a[15] = (_Float16)a3.w;

    for (int ct = 0; ct < NCPAD / 16; ++ct) {       // uniform loop: EXEC full at WMMA
        int col = ct * 16 + r;                      // this lane's output column
        bool valid = (col < NOUTC);

        // B tile from LDS (padded with zero columns -> no clamp)
        v16h b = *(const v16h*)(shW + (size_t)col * 32 + half * 16);

        v8f c = {};
        v8f dd = __builtin_amdgcn_wmma_f32_16x16x32_f16(false, a, false, b, (short)0, c, false, false);

        int p = col >> 1, s = col & 1;
        bool writer = ((col & 1) == 0) && valid;
        int iu = 0, ju = 0;
        if (writer) {
            int pr = sh_pair[p];
            iu = pr >> 8;
            ju = pr & 255;
        }
        float bcol = valid ? bias[col] : 0.0f;

#pragma unroll
        for (int i = 0; i < 8; ++i) {
            int  m = i + half * 8;
            long g = g0 + m;
            float z = dd[i] + bcol + (valid ? gum[g * (long)NOUTC + col] : 0.0f);
            float zn = __shfl_down(z, 1, 32);       // all lanes execute the shuffle
            if (writer) {
                float val = (z >= zn) ? 1.0f : 0.0f;   // argmax ties -> index 0
                adj[g * 2500 + iu * 50 + ju] = val;
                adj[g * 2500 + ju * 50 + iu] = val;
            }
        }
    }
}

// ---------------------------------------------------------------------------
extern "C" void kernel_launch(void* const* d_in, const int* in_sizes, int n_in,
                              void* d_out, int out_size, void* d_ws, size_t ws_size,
                              hipStream_t stream)
{
    const float* x        = (const float*)d_in[0];   // [N,10]
    const int*   ei       = (const int*)  d_in[1];   // [2,E]
    const float* gum      = (const float*)d_in[3];   // [B,1225,2]
    const float* enc0_W1  = (const float*)d_in[4];
    const float* enc0_b1  = (const float*)d_in[5];
    const float* enc0_g   = (const float*)d_in[6];
    const float* enc0_be  = (const float*)d_in[7];
    const float* enc0_W2  = (const float*)d_in[8];
    const float* enc0_b2  = (const float*)d_in[9];
    const float* enc1_W1  = (const float*)d_in[10];
    const float* enc1_b1  = (const float*)d_in[11];
    const float* enc1_g   = (const float*)d_in[12];
    const float* enc1_be  = (const float*)d_in[13];
    const float* enc1_W2  = (const float*)d_in[14];
    const float* enc1_b2  = (const float*)d_in[15];
    const float* dec_W0   = (const float*)d_in[16];
    const float* dec_b0   = (const float*)d_in[17];
    const float* dec_W1   = (const float*)d_in[18];
    const float* dec_b1   = (const float*)d_in[19];
    float* adj = (float*)d_out;

    const size_t NH = (size_t)NNODES * 32;
    float* agg    = (float*)d_ws;
    float* bufP   = agg  + NH;
    float* bufH   = bufP + NH;
    float* xg     = bufH + NH;
    float* dgr    = xg   + (size_t)NGRAPH * 32;
    float* stats0 = dgr  + (size_t)NGRAPH * 32;
    float* stats1 = stats0 + 64;
    float* bnp0   = stats1 + 64;
    float* bnp1   = bnp0 + 64;
    _Float16* W1h = (_Float16*)(bnp1 + 64);          // [2464][32] f16

    const int EB1 = NEDGES / 256;          // 25600  (exact)
    const int EB2 = (NEDGES * 4) / 256;    // 102400 (exact)
    const int NB  = NNODES / 128;          // 6400   (exact, full waves -> EXEC all 1s)
    const int GB  = NGRAPH / 128;          // 128

    (void)hipMemsetAsync(stats0, 0, 128 * sizeof(float), stream);
    k_cvtW1<<<(NCPAD * 32) / 256, 256, 0, stream>>>(dec_W1, W1h);

    // ---- Layer 1 ----
    (void)hipMemsetAsync(agg, 0, NH * sizeof(float), stream);
    k_scatter10<<<EB1, 256, 0, stream>>>(ei, x, agg);
    k_fused_gemm<10, true,  false, true ><<<NB, 256, 0, stream>>>(
        x, 10, agg, nullptr, enc0_W1, enc0_b1, bufP, stats0);
    k_bnfin<<<1, 32, 0, stream>>>(stats0, enc0_g, enc0_be, bnp0, 1.0f / (float)NNODES);
    k_fused_gemm<32, false, true,  false><<<NB, 256, 0, stream>>>(
        bufP, 32, nullptr, bnp0, enc0_W2, enc0_b2, bufH, nullptr);

    // ---- Layer 2 ----
    (void)hipMemsetAsync(agg, 0, NH * sizeof(float), stream);
    k_scatter32<<<EB2, 256, 0, stream>>>(ei, bufH, agg);
    k_fused_gemm<32, true,  false, true ><<<NB, 256, 0, stream>>>(
        bufH, 32, agg, nullptr, enc1_W1, enc1_b1, bufP, stats1);
    k_bnfin<<<1, 32, 0, stream>>>(stats1, enc1_g, enc1_be, bnp1, 1.0f / (float)NNODES);
    k_fused_gemm<32, false, true,  false><<<NB, 256, 0, stream>>>(
        bufP, 32, nullptr, bnp1, enc1_W2, enc1_b2, bufH, nullptr);

    // ---- Pool + decoder ----
    k_graphsum<<<(NGRAPH * 32) / 256, 256, 0, stream>>>(bufH, xg);
    k_fused_gemm<32, false, false, false><<<GB, 256, 0, stream>>>(
        xg, 32, nullptr, nullptr, dec_W0, dec_b0, dgr, nullptr);

    (void)hipMemsetAsync(adj, 0, (size_t)NGRAPH * 2500 * sizeof(float), stream);
    k_decode<<<GB, 256, 0, stream>>>(dgr, W1h, dec_b1, gum, adj);
}
